// PLinear_18777597018304
// MI455X (gfx1250) — matrix operand order repaired and checked by reading, not compiled, not run on hardware
//
#include <hip/hip_runtime.h>
#include <hip/hip_bf16.h>

// ---------------------------------------------------------------------------
// PLinear fused kernel for MI455X (gfx1250, wave32, WMMA + async-to-LDS).
//   out[256,8192] = x[256,8192] @ sample(mean,sigma)[8192,8192]^T + bias
// Pass 1: x f32 -> bf16 once into d_ws (4MB, L2-resident).
// Pass 2: one WG = all 256 rows x 64 output cols; W sampled exactly once
//   globally (512MB HBM = roofline min). Double-buffered LDS; x tile streamed
//   via global_load_async_to_lds_b128 (ASYNCcnt); W tile sampled in VALU while
//   the v_wmma_f32_16x16x32_bf16 burst of the previous tile executes.
// ---------------------------------------------------------------------------

#define B_ROWS 256
#define IN_DIM 8192
#define OUT_DIM 8192
#define KSTEP 32
#define N_TILE 64          // output columns per workgroup
#define LDS_STRIDE 40      // bf16/row (80B: 16B-aligned b128s, bank-clean)
#define AS_BUF (B_ROWS * LDS_STRIDE)
#define BS_BUF (N_TILE * LDS_STRIDE)

typedef __bf16 v4bf  __attribute__((ext_vector_type(4)));
typedef __bf16 v8bf  __attribute__((ext_vector_type(8)));
typedef __bf16 v16bf __attribute__((ext_vector_type(16)));
typedef float  v8f   __attribute__((ext_vector_type(8)));

__device__ __forceinline__ unsigned mix32(unsigned h) {
    h ^= h >> 16; h *= 0x85ebca6bu;
    h ^= h >> 13; h *= 0xc2b2ae35u;
    h ^= h >> 16;
    return h;
}

// Deterministic per-element weight sample. Normal draw via Irwin-Hall(4)
// (4x16-bit uniforms, pure FMA); raw v_log/v_sqrt/v_rcp for the pdf inversion.
__device__ __forceinline__ float sample_w(float mean, float sigma, unsigned idx) {
    unsigned h1 = mix32(idx * 0x9E3779B9u + 0x42424242u);
    unsigned h2 = mix32(h1 ^ (idx * 0x85EBCA77u + 0xC2B2AE3Du));
    float s01 = (float)(h1 & 0xffffu) + (float)(h1 >> 16);
    float s23 = (float)(h2 & 0xffffu) + (float)(h2 >> 16);
    float nrm = __builtin_fmaf(s01 + s23, 0x1.0p-16f, -2.0f) * 1.7320508f;

    float s2     = sigma * sigma;
    float inv_mu = sigma * 2.50662827463f;                   // sqrt(2*pi)*sigma
    float mu_y   = __builtin_amdgcn_rcpf(inv_mu);            // 1/sqrt(2*pi*s2)
    float sig_y  = __builtin_amdgcn_sqrtf(mu_y * 0.39346934028f);
    float y      = mu_y + sig_y * nrm;
    y = (y <= mu_y) ? y : (y - mu_y);                        // fold above mode
    float arg  = __builtin_fabsf(y * inv_mu);                // y*sqrt(2*pi*s2)
    float l2   = __builtin_amdgcn_logf(arg);                 // v_log_f32
    float disp = __builtin_amdgcn_sqrtf(__builtin_fabsf(-1.38629436112f * s2 * l2));
    float sgn  = ((h1 ^ h2) & 0x80000000u) ? 1.0f : -1.0f;
    return __builtin_fmaf(disp, sgn, mean);
}

// ---- Pass 1: convert x to bf16 once -----------------------------------
__global__ __launch_bounds__(256)
void cvt_x_kernel(const float* __restrict__ x, __bf16* __restrict__ xbf) {
    int i = (blockIdx.x * 256 + threadIdx.x) * 4;
    float4 v = *reinterpret_cast<const float4*>(x + i);
    v4bf o = { (__bf16)v.x, (__bf16)v.y, (__bf16)v.z, (__bf16)v.w };
    *reinterpret_cast<v4bf*>(xbf + i) = o;
}

// x tile [256 x 32] bf16: async DMA straight into padded LDS, no VGPR staging.
__device__ __forceinline__ void stage_x_async(const __bf16* __restrict__ xbf,
                                              __bf16* __restrict__ as,
                                              int t, int k0) {
    #pragma unroll
    for (int it = 0; it < 4; ++it) {
        int e16 = it * 256 + t;                  // 1024 16B segments
        int row = e16 >> 2;
        int seg = e16 & 3;
        const __bf16* gsrc = xbf + row * IN_DIM + k0 + seg * 8;
        unsigned ldsoff = (unsigned)(uintptr_t)&as[row * LDS_STRIDE + seg * 8];
        asm volatile("global_load_async_to_lds_b128 %0, %1, off"
                     :: "v"(ldsoff), "v"(gsrc) : "memory");
    }
}

// Fallback x staging from f32 (used when d_ws is too small for xbf)
__device__ __forceinline__ void stage_x_direct(const float* __restrict__ x,
                                               __bf16* __restrict__ as,
                                               int t, int k0) {
    #pragma unroll
    for (int it = 0; it < 8; ++it) {
        int e4   = it * 256 + t;
        int row  = e4 >> 3;
        int col4 = e4 & 7;
        float4 v = *reinterpret_cast<const float4*>(x + row * IN_DIM + k0 + col4 * 4);
        v4bf o = { (__bf16)v.x, (__bf16)v.y, (__bf16)v.z, (__bf16)v.w };
        *reinterpret_cast<v4bf*>(&as[row * LDS_STRIDE + col4 * 4]) = o;
    }
}

// W tile [64 x 32] sampled from (mean, sigma)
__device__ __forceinline__ void stage_w(const float* __restrict__ mean,
                                        const float* __restrict__ sigma,
                                        __bf16* __restrict__ bs,
                                        int t, int n0, int k0, bool prefetch) {
    #pragma unroll
    for (int it = 0; it < 2; ++it) {
        int e4 = it * 256 + t;                   // 512 float4 total
        int o  = e4 >> 3;
        int k4 = e4 & 7;
        unsigned gbase = (unsigned)(n0 + o) * (unsigned)IN_DIM + (unsigned)(k0 + k4 * 4);
        float4 m4 = *reinterpret_cast<const float4*>(mean  + gbase);
        float4 s4 = *reinterpret_cast<const float4*>(sigma + gbase);
        if (prefetch) {                          // global_prefetch_b8 next tile
            __builtin_prefetch(mean  + gbase + KSTEP, 0, 1);
            __builtin_prefetch(sigma + gbase + KSTEP, 0, 1);
        }
        v4bf o4 = { (__bf16)sample_w(m4.x, s4.x, gbase + 0u),
                    (__bf16)sample_w(m4.y, s4.y, gbase + 1u),
                    (__bf16)sample_w(m4.z, s4.z, gbase + 2u),
                    (__bf16)sample_w(m4.w, s4.w, gbase + 3u) };
        *reinterpret_cast<v4bf*>(&bs[o * LDS_STRIDE + k4 * 4]) = o4;
    }
}

template<bool ASYNC>
__global__ __launch_bounds__(256)
void plinear_wmma_kernel(const float* __restrict__ x,
                         const __bf16* __restrict__ xbf,
                         const float* __restrict__ mean,
                         const float* __restrict__ sigma,
                         const float* __restrict__ bias,
                         float* __restrict__ out) {
    __shared__ __align__(16) __bf16 As[2 * AS_BUF];   // 40960 B
    __shared__ __align__(16) __bf16 Bs[2 * BS_BUF];   // 10240 B

    const int t    = threadIdx.x;
    const int wid  = t >> 5;
    const int lane = t & 31;
    const int lrow = lane & 15;
    const int kh   = lane >> 4;

    const int n0    = blockIdx.x * N_TILE;       // output column base
    const int mbase = (wid & 3) * 64;            // wave's 64 rows of x
    const int nbase = (wid >> 2) * 32;           // wave's 32 cols within tile

    v8f acc[4][2];
    #pragma unroll
    for (int mt = 0; mt < 4; ++mt)
        #pragma unroll
        for (int nt = 0; nt < 2; ++nt)
            acc[mt][nt] = (v8f){0.f,0.f,0.f,0.f,0.f,0.f,0.f,0.f};

    const int KT = IN_DIM / KSTEP;               // 256 K-steps

    // Prologue: stage tile 0 into buffer 0
    if (ASYNC) stage_x_async(xbf, As, t, 0);
    else       stage_x_direct(x, As, t, 0);
    stage_w(mean, sigma, Bs, t, n0, 0, true);

    for (int kt = 0; kt < KT; ++kt) {
        const int buf = kt & 1;
        const __bf16* as = As + buf * AS_BUF;
        const __bf16* bs = Bs + buf * BS_BUF;

        // Drain this thread's async DMAs for tile kt, then publish via barrier
        if (ASYNC) asm volatile("s_wait_asynccnt 0x0" ::: "memory");
        __syncthreads();

        if (kt + 1 < KT) {
            __bf16* asn = As + (buf ^ 1) * AS_BUF;
            __bf16* bsn = Bs + (buf ^ 1) * BS_BUF;
            if (ASYNC) stage_x_async(xbf, asn, t, (kt + 1) * KSTEP);
            else       stage_x_direct(x, asn, t, (kt + 1) * KSTEP);
            stage_w(mean, sigma, bsn, t, n0, (kt + 1) * KSTEP, kt + 2 < KT);
        }

        // ---- WMMA burst: 8 x v_wmma_f32_16x16x32_bf16 per wave ----
        v16bf bfrag[2];
        #pragma unroll
        for (int nt = 0; nt < 2; ++nt) {
            int n = nbase + nt * 16 + lrow;
            const v8bf* p = reinterpret_cast<const v8bf*>(&bs[n * LDS_STRIDE + kh * 16]);
            v8bf lo = p[0], hi = p[1];
            bfrag[nt] = __builtin_shufflevector(lo, hi,
                0,1,2,3,4,5,6,7,8,9,10,11,12,13,14,15);
        }
        #pragma unroll
        for (int mt = 0; mt < 4; ++mt) {
            int r = mbase + mt * 16 + lrow;
            v8bf lo = *reinterpret_cast<const v8bf*>(&as[r * LDS_STRIDE + kh * 16]);
            v8bf hi = *reinterpret_cast<const v8bf*>(&as[r * LDS_STRIDE + kh * 16 + 8]);
            v16bf afrag = __builtin_shufflevector(lo, hi,
                0,1,2,3,4,5,6,7,8,9,10,11,12,13,14,15);
            #pragma unroll
            for (int nt = 0; nt < 2; ++nt) {
                acc[mt][nt] = __builtin_amdgcn_wmma_f32_16x16x32_bf16(
                    false, afrag, false, bfrag[nt],
                    (short)0, acc[mt][nt], false, false);
            }
        }
    }

    // ---- epilogue: bias add + store (D layout: VGPR r -> M = kh*8 + r) ----
    #pragma unroll
    for (int nt = 0; nt < 2; ++nt) {
        int col = n0 + nbase + nt * 16 + lrow;
        float bv = bias[col];
        #pragma unroll
        for (int mt = 0; mt < 4; ++mt) {
            #pragma unroll
            for (int r = 0; r < 8; ++r) {
                int row = mbase + mt * 16 + kh * 8 + r;
                out[row * OUT_DIM + col] = acc[mt][nt][r] + bv;
            }
        }
    }
}

extern "C" void kernel_launch(void* const* d_in, const int* in_sizes, int n_in,
                              void* d_out, int out_size, void* d_ws, size_t ws_size,
                              hipStream_t stream) {
    (void)in_sizes; (void)n_in; (void)out_size;
    const float* x     = (const float*)d_in[0];
    const float* mean  = (const float*)d_in[1];
    const float* sigma = (const float*)d_in[2];
    const float* bias  = (const float*)d_in[3];
    float* out = (float*)d_out;

    const size_t xbf_bytes = (size_t)B_ROWS * IN_DIM * sizeof(__bf16); // 4MB
    if (ws_size >= xbf_bytes) {
        __bf16* xbf = (__bf16*)d_ws;
        cvt_x_kernel<<<dim3((B_ROWS * IN_DIM) / 1024), dim3(256), 0, stream>>>(x, xbf);
        plinear_wmma_kernel<true><<<dim3(OUT_DIM / N_TILE), dim3(256), 0, stream>>>(
            x, xbf, mean, sigma, bias, out);
    } else {
        plinear_wmma_kernel<false><<<dim3(OUT_DIM / N_TILE), dim3(256), 0, stream>>>(
            x, nullptr, mean, sigma, bias, out);
    }
}
